// GlobalPointer_34144990003256
// MI455X (gfx1250) — compile-verified, matrix operand order checked by reading, not run on hardware
//
#include <hip/hip_runtime.h>
#include <hip/hip_bf16.h>

typedef __attribute__((ext_vector_type(16))) __bf16 v16bf;
typedef __attribute__((ext_vector_type(8)))  float  v8f;

union FragAB { uint4 q[2]; v16bf v; };

__device__ __forceinline__ unsigned short f32_to_bf16(float f) {
    unsigned int u = __float_as_uint(f);
    u += 0x7FFFu + ((u >> 16) & 1u);           // round-to-nearest-even
    return (unsigned short)(u >> 16);
}

// ---- CDNA5 async memory->LDS DMA (GLOBAL_LOAD_ASYNC_TO_LDS_B128, ASYNCcnt) ----
// LDS operand: low 32 bits of the flat shared-aperture address == LDS byte offset.
__device__ __forceinline__ void async_copy16(const void* g, void* l) {
    unsigned      lds = (unsigned)(uintptr_t)l;
    unsigned long long ga = (unsigned long long)(uintptr_t)g;
    asm volatile("global_load_async_to_lds_b128 %0, %1, off"
                 :: "v"(lds), "v"(ga) : "memory");
}
__device__ __forceinline__ void wait_async0() {
    asm volatile("s_wait_asynccnt 0x0" ::: "memory");
}

// ---------------- Kernel 0a: hidden f32 -> bf16 (row-major [m][k]) -------------
__global__ __launch_bounds__(256) void cvt_hidden(const float* __restrict__ src,
                                                  unsigned short* __restrict__ dst, int n) {
    int i = blockIdx.x * 256 + threadIdx.x;
    if (i < n) dst[i] = f32_to_bf16(src[i]);
}

// ------------- Kernel 0b: W f32 [k=768][n=1024] -> Wt bf16 [n][k] --------------
__global__ __launch_bounds__(256) void cvt_w(const float* __restrict__ w,
                                             unsigned short* __restrict__ wt) {
    int i = blockIdx.x * 256 + threadIdx.x;   // i over 768*1024
    int k = i >> 10, n = i & 1023;
    wt[n * 768 + k] = f32_to_bf16(w[i]);
}

// ---------------- Kernel 1: proj = hidden @ W + b, then RoPE -> Q/K bf16 -------
// grid: (64 m-tiles of 128, 8 heads), block 256 (8 waves as 4Mx2N, wave = 32x64)
__global__ __launch_bounds__(256) void proj_rope(
    const unsigned short* __restrict__ hbf,   // [8192][768] bf16
    const unsigned short* __restrict__ wtb,   // [1024][768] bf16 (W transposed)
    const float* __restrict__ bias,           // [1024]
    unsigned short* __restrict__ qb,          // [b][e][s][64] bf16
    unsigned short* __restrict__ kb)          // [b][e][s][64] bf16
{
    __shared__ __align__(16) unsigned short As[2][128][32];   // double buffered
    __shared__ __align__(16) unsigned short Bs[2][128][32];

    const int m0   = blockIdx.x * 128;
    const int e    = blockIdx.y;
    const int t    = threadIdx.x;
    const int wave = t >> 5, lane = t & 31;
    const int waveM = wave & 3, waveN = wave >> 2;            // 4x2
    const int lm = lane & 15, hi = lane >> 4;

    const int row = t >> 1, seg = t & 1;                      // staging assignment
    const unsigned short* gA = &hbf[(size_t)(m0 + row) * 768 + seg * 16];
    const unsigned short* gB = &wtb[(size_t)(e * 128 + row) * 768 + seg * 16];

    v8f acc[2][4];
    #pragma unroll
    for (int i = 0; i < 2; ++i)
        #pragma unroll
        for (int j = 0; j < 4; ++j)
            acc[i][j] = (v8f){0.f,0.f,0.f,0.f,0.f,0.f,0.f,0.f};

    // stage tile 0 into buffer 0
    async_copy16(gA,     &As[0][row][seg * 16]);
    async_copy16(gA + 8, &As[0][row][seg * 16 + 8]);
    async_copy16(gB,     &Bs[0][row][seg * 16]);
    async_copy16(gB + 8, &Bs[0][row][seg * 16 + 8]);
    wait_async0();
    __syncthreads();

    for (int kt = 0; kt < 24; ++kt) {
        const int cur = kt & 1;
        if (kt + 1 < 24) {                                    // async-prefetch next tile
            const int nb = (kt + 1) & 1, k0n = (kt + 1) * 32;
            async_copy16(gA + k0n,     &As[nb][row][seg * 16]);
            async_copy16(gA + k0n + 8, &As[nb][row][seg * 16 + 8]);
            async_copy16(gB + k0n,     &Bs[nb][row][seg * 16]);
            async_copy16(gB + k0n + 8, &Bs[nb][row][seg * 16 + 8]);
        }
        if (kt + 2 < 24) {                                    // warm L2 one tile further
            __builtin_prefetch(gA + (kt + 2) * 32, 0, 1);
            __builtin_prefetch(gB + (kt + 2) * 32, 0, 1);
        }

        FragAB a[2];
        #pragma unroll
        for (int i = 0; i < 2; ++i) {
            int r = waveM * 32 + i * 16 + lm;
            a[i].q[0] = *(const uint4*)&As[cur][r][hi * 8];
            a[i].q[1] = *(const uint4*)&As[cur][r][16 + hi * 8];
        }
        #pragma unroll
        for (int j = 0; j < 4; ++j) {
            int c = waveN * 64 + j * 16 + lm;
            FragAB bf;
            bf.q[0] = *(const uint4*)&Bs[cur][c][hi * 8];
            bf.q[1] = *(const uint4*)&Bs[cur][c][16 + hi * 8];
            #pragma unroll
            for (int i = 0; i < 2; ++i)
                acc[i][j] = __builtin_amdgcn_wmma_f32_16x16x32_bf16(
                    false, a[i].v, false, bf.v, (short)0, acc[i][j], false, false);
        }

        wait_async0();          // next tile's DMA done (ASYNCcnt not covered by barrier)
        __syncthreads();
    }

    // epilogue: bias + interleaved RoPE via lane-pair shuffle, store bf16
    #pragma unroll
    for (int i = 0; i < 2; ++i) {
        #pragma unroll
        for (int j = 0; j < 4; ++j) {
            int inner = waveN * 64 + j * 16 + lm;      // 0..127 within head
            int col   = e * 128 + inner;
            float bv  = bias[col];
            int  d_in = (inner < 64) ? inner : inner - 64;
            int  p    = d_in >> 1;
            // inv_freq = 10000^(-2p/64) = exp(-ln(10000)/32 * p)
            float invf = __expf(-0.28782313662425572f * (float)p);
            #pragma unroll
            for (int r = 0; r < 8; ++r) {
                int m = m0 + waveM * 32 + i * 16 + r + hi * 8;
                int b = m >> 9, s = m & 511;
                float vb   = acc[i][j][r] + bv;
                float pair = __shfl_xor(vb, 1, 32);    // element at d ^ 1
                float ang  = (float)s * invf;
                float sp = __sinf(ang), cp = __cosf(ang);
                float outv = (d_in & 1) ? (vb * cp + pair * sp)
                                        : (vb * cp - pair * sp);
                unsigned short ob = f32_to_bf16(outv);
                size_t idx = ((size_t)((b * 8 + e) * 512 + s)) * 64 + d_in;
                if (inner < 64) qb[idx] = ob; else kb[idx] = ob;
            }
        }
    }
}

// ---------------- Kernel 2: logits = Q @ K^T, mask, causal(0,0), *0.125 --------
// grid: (4 m-tiles, 4 n-tiles, 128 bh), block 256 (8 waves as 4Mx2N)
__global__ __launch_bounds__(256) void logits_kernel(
    const unsigned short* __restrict__ qb,    // [bh][512][64] bf16
    const unsigned short* __restrict__ kb,    // [bh][512][64] bf16
    const int* __restrict__ am,               // [16][512]
    float* __restrict__ out)                  // [bh][512][512]
{
    __shared__ __align__(16) unsigned short Qs[128][64];
    __shared__ __align__(16) unsigned short Ks[128][64];

    const int m0 = blockIdx.x * 128;
    const int n0 = blockIdx.y * 128;
    const int bh = blockIdx.z;
    const int b  = bh >> 3;
    const int t  = threadIdx.x;
    const int wave = t >> 5, lane = t & 31;
    const int waveM = wave & 3, waveN = wave >> 2;
    const int lm = lane & 15, hi = lane >> 4;

    {   // stage full 128x64 Q and K tiles via async DMA (whole K dim = 64)
        int row = t >> 1, seg = t & 1;
        const unsigned short* gq = &qb[((size_t)bh * 512 + m0 + row) * 64 + seg * 32];
        const unsigned short* gk = &kb[((size_t)bh * 512 + n0 + row) * 64 + seg * 32];
        #pragma unroll
        for (int c = 0; c < 4; ++c) {
            async_copy16(gq + c * 8, &Qs[row][seg * 32 + c * 8]);
            async_copy16(gk + c * 8, &Ks[row][seg * 32 + c * 8]);
        }
    }
    wait_async0();
    __syncthreads();

    v8f acc[2][4];
    #pragma unroll
    for (int i = 0; i < 2; ++i)
        #pragma unroll
        for (int j = 0; j < 4; ++j)
            acc[i][j] = (v8f){0.f,0.f,0.f,0.f,0.f,0.f,0.f,0.f};

    #pragma unroll
    for (int k0 = 0; k0 < 64; k0 += 32) {
        FragAB a[2];
        #pragma unroll
        for (int i = 0; i < 2; ++i) {
            int r = waveM * 32 + i * 16 + lm;
            a[i].q[0] = *(const uint4*)&Qs[r][k0 + hi * 8];
            a[i].q[1] = *(const uint4*)&Qs[r][k0 + 16 + hi * 8];
        }
        #pragma unroll
        for (int j = 0; j < 4; ++j) {
            int c = waveN * 64 + j * 16 + lm;
            FragAB bf;
            bf.q[0] = *(const uint4*)&Ks[c][k0 + hi * 8];
            bf.q[1] = *(const uint4*)&Ks[c][k0 + 16 + hi * 8];
            #pragma unroll
            for (int i = 0; i < 2; ++i)
                acc[i][j] = __builtin_amdgcn_wmma_f32_16x16x32_bf16(
                    false, a[i].v, false, bf.v, (short)0, acc[i][j], false, false);
        }
    }

    // epilogue: padding mask on key n, strict-lower causal for (b,h)==(0,0), /8
    #pragma unroll
    for (int j = 0; j < 4; ++j) {
        int n = n0 + waveN * 64 + j * 16 + lm;
        float pm = (float)am[b * 512 + n];
        #pragma unroll
        for (int i = 0; i < 2; ++i) {
            #pragma unroll
            for (int r = 0; r < 8; ++r) {
                int m = m0 + waveM * 32 + i * 16 + r + hi * 8;
                float v = acc[i][j][r];
                v = v * pm - (1.0f - pm) * 1.0e12f;
                if (bh == 0 && m > n) v -= 1.0e12f;
                v *= 0.125f;
                out[((size_t)bh * 512 + m) * 512 + n] = v;
            }
        }
    }
}

extern "C" void kernel_launch(void* const* d_in, const int* in_sizes, int n_in,
                              void* d_out, int out_size, void* d_ws, size_t ws_size,
                              hipStream_t stream) {
    const float* hidden = (const float*)d_in[0];   // (16,512,768)
    const float* W      = (const float*)d_in[1];   // (768,1024)
    const float* bias   = (const float*)d_in[2];   // (1024,)
    const int*   amask  = (const int*)d_in[3];     // (16,512)
    float* out = (float*)d_out;                    // (16,8,512,512)

    unsigned short* hbf  = (unsigned short*)d_ws;                  // 8192*768
    unsigned short* wtb  = hbf + (size_t)8192 * 768;               // 1024*768
    unsigned short* qbuf = wtb + (size_t)1024 * 768;               // 128*512*64
    unsigned short* kbuf = qbuf + (size_t)128 * 512 * 64;          // 128*512*64

    cvt_hidden<<<(8192 * 768) / 256, 256, 0, stream>>>(hidden, hbf, 8192 * 768);
    cvt_w<<<(768 * 1024) / 256, 256, 0, stream>>>(W, wtb);
    proj_rope<<<dim3(64, 8), 256, 0, stream>>>(hbf, wtb, bias, qbuf, kbuf);
    logits_kernel<<<dim3(4, 4, 128), 256, 0, stream>>>(qbuf, kbuf, amask, out);
}
